// MGCN_3745211482329
// MI455X (gfx1250) — matrix operand order repaired and checked by reading, not compiled, not run on hardware
//
#include <hip/hip_runtime.h>
#include <hip/hip_bf16.h>
#include <stdint.h>

typedef __attribute__((ext_vector_type(16))) _Float16 v16h;
typedef __attribute__((ext_vector_type(8)))  _Float16 v8h;
typedef __attribute__((ext_vector_type(8)))  float    v8f;

typedef unsigned u32x4 __attribute__((ext_vector_type(4)));
typedef int      i32x8 __attribute__((ext_vector_type(8)));
typedef int      i32x4 __attribute__((ext_vector_type(4)));

#define NNODES 20000
#define NEDGES 32768
#define DIM    128
#define NREL   474
#define NRELP  480      // NREL padded to a multiple of 16 (pad rows never consumed)
#define NBASIS 64

// LDS layout for k_msg: W_r staged with 1-dword row pad (65 dwords = 260 B per row
// of 128 halves) -> bank-conflict-free column reads. A tile padded to 272 B rows
// (16B aligned) -> conflict-free ds_read_b128.
#define W_ROW_B   260
#define W_LDS_B   (128 * W_ROW_B)          // 33280
#define A_ROW_B   272
#define A_LDS_OFF W_LDS_B                  // 16B aligned (33280 % 16 == 0)
#define A_LDS_B   (16 * A_ROW_B)           // 4352
#define MSG_LDS_B (W_LDS_B + A_LDS_B)      // 37632

#if defined(__has_builtin)
#if __has_builtin(__builtin_amdgcn_tensor_load_to_lds) && \
    __has_builtin(__builtin_amdgcn_s_wait_tensorcnt)
#define HAVE_TDM 1
#endif
#if __has_builtin(__builtin_amdgcn_global_load_async_to_lds_b128)
#define HAVE_ASYNC 1
#endif
#endif
#ifndef HAVE_TDM
#define HAVE_TDM 0
#endif
#ifndef HAVE_ASYNC
#define HAVE_ASYNC 0
#endif

// ---------- small helpers ----------

__device__ __forceinline__ unsigned encodeF(float f) {
    unsigned b = __float_as_uint(f);
    return (b & 0x80000000u) ? ~b : (b | 0x80000000u);
}
__device__ __forceinline__ float decodeF(unsigned k) {
    unsigned b = (k & 0x80000000u) ? (k & 0x7FFFFFFFu) : ~k;
    return __uint_as_float(b);
}

// A fragment (16-bit A 16x32): per-lane halves are K-runs {kb..kb+15}.
__device__ __forceinline__ v16h load_a_runs(const _Float16* p) {
    v8h lo = *(const v8h*)p;
    v8h hi = *(const v8h*)(p + 8);
    return __builtin_shufflevector(lo, hi, 0, 1, 2, 3, 4, 5, 6, 7,
                                   8, 9, 10, 11, 12, 13, 14, 15);
}

// B fragment from K-contiguous (transposed) storage: 16 consecutive halves at p.
__device__ __forceinline__ v16h load_contig16(const _Float16* p) {
    v8h lo = *(const v8h*)p;
    v8h hi = *(const v8h*)(p + 8);
    return __builtin_shufflevector(lo, hi, 0, 1, 2, 3, 4, 5, 6, 7,
                                   8, 9, 10, 11, 12, 13, 14, 15);
}

// 16-byte A-tile staging: async global->LDS when available, else load+ds_store.
__device__ __forceinline__ void stageA16(char* ldsDst, const char* gsrc) {
#if HAVE_ASYNC
    __builtin_amdgcn_global_load_async_to_lds_b128(
        (__attribute__((address_space(1))) i32x4*)(uintptr_t)gsrc,
        (__attribute__((address_space(3))) i32x4*)(unsigned)(uintptr_t)ldsDst,
        0, 0);
#else
    *(uint4*)ldsDst = *(const uint4*)gsrc;
#endif
}
__device__ __forceinline__ void stageWaitAsync() {
#if HAVE_ASYNC
#if defined(__has_builtin) && __has_builtin(__builtin_amdgcn_s_wait_asynccnt)
    __builtin_amdgcn_s_wait_asynccnt(0);
#else
    asm volatile("s_wait_asynccnt 0x0" ::: "memory");
#endif
#endif
}

#if HAVE_TDM
// 1-D TDM copy: nElems (2-byte) elements global->LDS with a 1-dword pad every
// 64 dwords (so each 128-half row lands on a 65-dword LDS stride).
__device__ __forceinline__ void tdm_load_w(unsigned ldsOff, const void* gptr,
                                           unsigned nElems) {
    unsigned long long ga = (unsigned long long)(uintptr_t)gptr;
    u32x4 g0;
    g0[0] = 1u;                                   // count=1, user descriptor
    g0[1] = ldsOff;                               // lds_addr
    g0[2] = (unsigned)(ga & 0xffffffffu);         // global_addr[31:0]
    g0[3] = (unsigned)((ga >> 32) & 0x01ffffffu)  // global_addr[56:32]
            | (2u << 30);                         // type=2 (image)
    i32x8 g1;
    g1[0] = (int)((1u << 16)                      // data_size = 2 bytes
                  | (1u << 20)                    // pad_enable
                  | (5u << 22));                  // pad_interval: 64 dwords
    g1[1] = (int)(nElems << 16);                  // tensor_dim0[15:0] in [31:16]
    g1[2] = (int)((nElems >> 16) & 0xffffu) | (1 << 16); // dim0 hi | tensor_dim1=1
    g1[3] = (int)(nElems << 16);                  // tile_dim0 in [31:16]
    g1[4] = 0;
    g1[5] = (int)nElems;                          // tensor_dim0_stride
    g1[6] = 0;
    g1[7] = 0;
    i32x4 z4 = {0, 0, 0, 0};
#if __has_include(<hip/amd_detail/amd_gfx1250_TDM.h>)
    i32x8 z8 = {0, 0, 0, 0, 0, 0, 0, 0};
    __builtin_amdgcn_tensor_load_to_lds(g0, g1, z4, z4, z8, 0);
#else
    __builtin_amdgcn_tensor_load_to_lds(g0, g1, z4, z4, 0);
#endif
}
#endif

// ---------- utility kernels ----------

__global__ void k_zero_u32(unsigned* p, int n) {
    int i = blockIdx.x * blockDim.x + threadIdx.x;
    if (i < n) p[i] = 0u;
}

__global__ void k_cast_f16(const float* __restrict__ in, _Float16* __restrict__ out, int n) {
    int i = blockIdx.x * blockDim.x + threadIdx.x;
    if (i < n) out[i] = (_Float16)in[i];
}

// cast + transpose: src [K][C] f32 row-major -> dstT [C][K] f16
__global__ void k_tcast(const float* __restrict__ src, _Float16* __restrict__ dstT,
                        int K, int C) {
    int i = blockIdx.x * blockDim.x + threadIdx.x;
    if (i >= K * C) return;
    int k = i / C, c = i - k * C;          // reads coalesced along c
    dstT[(size_t)c * K + k] = (_Float16)src[i];
}

__global__ void k_gather_x(const int* __restrict__ entity, const float* __restrict__ emb,
                           float* __restrict__ xf, _Float16* __restrict__ xh) {
    int i = blockIdx.x * blockDim.x + threadIdx.x;
    if (i >= NNODES * DIM) return;
    int row = i >> 7, col = i & 127;
    float v = emb[(size_t)entity[row] * DIM + col];
    xf[i] = v;
    xh[i] = (_Float16)v;
}

// ---------- edge bucketing by relation ----------

__global__ void k_count(const int* __restrict__ etype, int* __restrict__ counts) {
    int e = blockIdx.x * blockDim.x + threadIdx.x;
    if (e < NEDGES) atomicAdd(&counts[etype[e]], 1);
}

__global__ void k_scan(const int* __restrict__ counts, int* __restrict__ offsets,
                       int* __restrict__ cursor) {
    __shared__ int s[512];
    int t = threadIdx.x;
    int c = (t < NREL) ? counts[t] : 0;
    s[t] = c;
    __syncthreads();
    for (int o = 1; o < 512; o <<= 1) {
        int v = (t >= o) ? s[t - o] : 0;
        __syncthreads();
        s[t] += v;
        __syncthreads();
    }
    if (t < NREL) {
        int excl = s[t] - c;
        offsets[t] = excl;
        cursor[t]  = excl;
    }
}

__global__ void k_scatter(const int* __restrict__ etype, int* __restrict__ cursor,
                          int* __restrict__ sortedEdge) {
    int e = blockIdx.x * blockDim.x + threadIdx.x;
    if (e >= NEDGES) return;
    int pos = atomicAdd(&cursor[etype[e]], 1);
    sortedEdge[pos] = e;
}

// ---------- attention logits + segment softmax ----------

__global__ void k_alpha(const int* __restrict__ src, const int* __restrict__ dst,
                        const int* __restrict__ etype, const float* __restrict__ xf,
                        const float* __restrict__ weight, float* __restrict__ alphaRaw,
                        unsigned* __restrict__ nodeKey) {
    int e = (blockIdx.x * blockDim.x + threadIdx.x) >> 5;
    int lane = threadIdx.x & 31;
    if (e >= NEDGES) return;
    const float* xi = xf + (size_t)dst[e] * DIM;
    const float* xj = xf + (size_t)src[e] * DIM;
    const float* w  = weight + (size_t)etype[e] * DIM;
    float sum = 0.f;
    #pragma unroll
    for (int k = 0; k < 4; ++k) {
        int d = lane + k * 32;
        sum += xi[d] * w[d] * xj[d];
    }
    #pragma unroll
    for (int o = 16; o > 0; o >>= 1) sum += __shfl_xor(sum, o, 32);
    if (lane == 0) {
        alphaRaw[e] = sum;
        atomicMax(&nodeKey[dst[e]], encodeF(sum));
    }
}

__global__ void k_expsum(const int* __restrict__ dst, const float* __restrict__ alphaRaw,
                         const unsigned* __restrict__ nodeKey, float* __restrict__ aBuf,
                         float* __restrict__ denom) {
    int e = blockIdx.x * blockDim.x + threadIdx.x;
    if (e >= NEDGES) return;
    int d = dst[e];
    float m = decodeF(nodeKey[d]);
    if (!(fabsf(m) < 3.0e38f)) m = 0.f;
    float a = __expf(alphaRaw[e] - m);
    aBuf[e] = a;
    atomicAdd(&denom[d], a);
}

__global__ void k_scale(const int* __restrict__ dst, const float* __restrict__ aBuf,
                        const float* __restrict__ denom, float* __restrict__ scale) {
    int e = blockIdx.x * blockDim.x + threadIdx.x;
    if (e >= NEDGES) return;
    scale[e] = aBuf[e] / denom[dst[e]];
}

// ---------- WMMA kernels ----------

// W[r] = sum_b att[r,b] * basis[b] : [480(pad),64] x [64,16384].
// Relation dim padded to 480 so no row guards (pad rows written, never read).
__global__ void k_relW(const _Float16* __restrict__ atth, const _Float16* __restrict__ basisT,
                       _Float16* __restrict__ W) {
    int rowBase = blockIdx.x * 16;
    int colBase = blockIdx.y * 16;
    int lane = threadIdx.x & 31;
    bool hiL = lane >= 16;
    int lm = lane & 15;

    const _Float16* bcol = basisT + (size_t)(colBase + lm) * NBASIS;
    const _Float16* ap0  = atth + (size_t)(rowBase + lm) * NBASIS + (hiL ? 16 : 0);

    v8f acc = {};
    #pragma unroll
    for (int ks = 0; ks < 2; ++ks) {
        v16h a = load_a_runs(ap0 + ks * 32);
        v16h b = load_contig16(bcol + ks * 32 + (hiL ? 16 : 0));
        acc = __builtin_amdgcn_wmma_f32_16x16x32_f16(false, a, false, b, (short)0, acc,
                                                     false, false);
    }
    #pragma unroll
    for (int g = 0; g < 8; ++g) {
        int r = rowBase + g + (hiL ? 8 : 0);
        W[(size_t)r * (DIM * DIM) + colBase + lm] = (_Float16)acc[g];
    }
}

// Per-relation edge-message GEMM. One 256-thread block (8 waves) per relation:
// W_r staged to LDS once (TDM when available); per-chunk gathered x_j rows staged
// to LDS (async-to-LDS when available); wave w computes column tile w.
__global__ void k_msg(const int* __restrict__ sortedEdge, const int* __restrict__ offsets,
                      const int* __restrict__ counts, const int* __restrict__ src,
                      const int* __restrict__ dst, const _Float16* __restrict__ xh,
                      const _Float16* __restrict__ W, const float* __restrict__ scale,
                      float* __restrict__ out) {
    extern __shared__ char smem[];
    int r = blockIdx.x;
    int cnt = counts[r];
    if (cnt == 0) return;
    int base = offsets[r];
    int tid  = threadIdx.x;
    int wv   = tid >> 5;
    int lane = tid & 31;
    bool hiL = lane >= 16;
    int lm = lane & 15;
    int colBase = wv * 16;

    const _Float16* Wr = W + (size_t)r * (DIM * DIM);

    // ---- stage W_r into LDS (padded 65-dword row stride) ----
#if HAVE_TDM
    if (wv == 0) {
        tdm_load_w(0u, Wr, DIM * DIM);
        __builtin_amdgcn_s_wait_tensorcnt(0);
    }
#else
    {
        const unsigned* Wg = (const unsigned*)Wr;
        unsigned* Wl = (unsigned*)smem;
        for (int idx = tid; idx < (DIM * DIM) / 2; idx += 256) {
            int row = idx >> 6, c = idx & 63;
            Wl[row * 65 + c] = Wg[idx];
        }
    }
#endif
    __syncthreads();

    // ---- per-wave B fragments from LDS (bank-conflict-free column reads) ----
    v16h bf[4];
    #pragma unroll
    for (int ks = 0; ks < 4; ++ks) {
        int k0 = ks * 32 + (hiL ? 16 : 0);
        v16h b = {};
        #pragma unroll
        for (int i = 0; i < 16; ++i)
            b[i] = *(const _Float16*)(smem + (size_t)(k0 + i) * W_ROW_B
                                           + (colBase + lm) * 2);
        bf[ks] = b;
    }

    char* ldsA = smem + A_LDS_OFF;
    for (int chunk = 0; chunk * 16 < cnt; ++chunk) {
        __syncthreads();  // previous iteration's A reads complete
        {
            int rrow = tid >> 4, cch = tid & 15;
            int eIdx = chunk * 16 + rrow;
            char* dstp = ldsA + rrow * A_ROW_B + cch * 16;
            if (eIdx < cnt) {
                int e = sortedEdge[base + eIdx];
                const char* srcp = (const char*)(xh + (size_t)src[e] * DIM) + cch * 16;
                stageA16(dstp, srcp);
            } else {
                *(uint4*)dstp = make_uint4(0u, 0u, 0u, 0u);
            }
        }
        stageWaitAsync();
        __syncthreads();

        const _Float16* rowPtr =
            (const _Float16*)(ldsA + lm * A_ROW_B) + (hiL ? 16 : 0);
        v8f acc = {};
        #pragma unroll
        for (int ks = 0; ks < 4; ++ks) {
            v16h a = load_a_runs(rowPtr + ks * 32);
            acc = __builtin_amdgcn_wmma_f32_16x16x32_f16(false, a, false, bf[ks], (short)0,
                                                         acc, false, false);
        }
        #pragma unroll
        for (int g = 0; g < 8; ++g) {
            int m = g + (hiL ? 8 : 0);
            int eIdx = chunk * 16 + m;
            if (eIdx < cnt) {
                int e = sortedEdge[base + eIdx];
                float val = acc[g] * scale[e];
                atomicAdd(&out[(size_t)dst[e] * DIM + colBase + lm], val);
            }
        }
    }
}

// out = out(msg) + x @ root + bias (+relu). B from rootT [128][128] (K-contiguous).
__global__ void k_root(const _Float16* __restrict__ xh, const _Float16* __restrict__ rootT,
                       const float* __restrict__ bias, float* __restrict__ out, int doRelu) {
    extern __shared__ char smem[];  // A tile: 16 rows x 272 B
    int tid  = threadIdx.x;
    int wv   = tid >> 5;
    int lane = tid & 31;
    bool hiL = lane >= 16;
    int lm = lane & 15;
    int colBase = wv * 16;

    const _Float16* bcol = rootT + (size_t)(colBase + lm) * DIM;
    v16h bf[4];
    #pragma unroll
    for (int ks = 0; ks < 4; ++ks)
        bf[ks] = load_contig16(bcol + ks * 32 + (hiL ? 16 : 0));
    float bcolv = bias[colBase + lm];

    int nTiles = (NNODES + 15) / 16;
    for (int tile = blockIdx.x; tile < nTiles; tile += gridDim.x) {
        int rowBase = tile * 16;
        __syncthreads();
        {
            int rrow = tid >> 4, cch = tid & 15;
            int row = rowBase + rrow;
            char* dstp = smem + rrow * A_ROW_B + cch * 16;
            if (row < NNODES) {
                const char* srcp = (const char*)(xh + (size_t)row * DIM) + cch * 16;
                stageA16(dstp, srcp);
            } else {
                *(uint4*)dstp = make_uint4(0u, 0u, 0u, 0u);
            }
        }
        stageWaitAsync();
        __syncthreads();

        const _Float16* rowPtr =
            (const _Float16*)(smem + lm * A_ROW_B) + (hiL ? 16 : 0);
        v8f acc = {};
        #pragma unroll
        for (int ks = 0; ks < 4; ++ks) {
            v16h a = load_a_runs(rowPtr + ks * 32);
            acc = __builtin_amdgcn_wmma_f32_16x16x32_f16(false, a, false, bf[ks], (short)0,
                                                         acc, false, false);
        }
        #pragma unroll
        for (int g = 0; g < 8; ++g) {
            int row = rowBase + g + (hiL ? 8 : 0);
            if (row < NNODES) {
                size_t idx = (size_t)row * DIM + colBase + lm;
                float v = out[idx] + acc[g] + bcolv;
                if (doRelu) v = fmaxf(v, 0.f);
                out[idx] = v;
            }
        }
    }
}

// ---------- host side ----------

extern "C" void kernel_launch(void* const* d_in, const int* in_sizes, int n_in,
                              void* d_out, int out_size, void* d_ws, size_t ws_size,
                              hipStream_t stream) {
    const int N = NNODES, E = NEDGES, D = DIM, R = NREL, Bb = NBASIS;

    const int*   entity = (const int*)d_in[0];
    const int*   eidx   = (const int*)d_in[1];
    const int*   src    = eidx;
    const int*   dst    = eidx + E;
    const int*   etype  = (const int*)d_in[2];
    const float* emb    = (const float*)d_in[3];
    const float* basis1 = (const float*)d_in[4];
    const float* att1   = (const float*)d_in[5];
    const float* wgt1   = (const float*)d_in[6];
    const float* root1  = (const float*)d_in[7];
    const float* bias1  = (const float*)d_in[8];
    const float* basis2 = (const float*)d_in[9];
    const float* att2   = (const float*)d_in[10];
    const float* wgt2   = (const float*)d_in[11];
    const float* root2  = (const float*)d_in[12];
    const float* bias2  = (const float*)d_in[13];

    char* wsBase = (char*)d_ws;
    size_t off = 0;
    auto alloc = [&](size_t bytes) -> void* {
        void* p = wsBase + off;
        off = (off + bytes + 255) & ~(size_t)255;
        return p;
    };
    float*     xf        = (float*)alloc((size_t)N * D * 4);
    float*     hbuf      = (float*)alloc((size_t)N * D * 4);
    _Float16*  xh        = (_Float16*)alloc((size_t)N * D * 2);
    _Float16*  Wf16      = (_Float16*)alloc((size_t)NRELP * D * D * 2);  // padded
    _Float16*  atth      = (_Float16*)alloc((size_t)NRELP * Bb * 2);     // padded
    _Float16*  basisT    = (_Float16*)alloc((size_t)Bb * D * D * 2);
    _Float16*  rootT     = (_Float16*)alloc((size_t)D * D * 2);
    float*     alphaRaw  = (float*)alloc((size_t)E * 4);
    float*     aBuf      = (float*)alloc((size_t)E * 4);
    float*     scaleBuf  = (float*)alloc((size_t)E * 4);
    unsigned*  nodeKey   = (unsigned*)alloc((size_t)N * 4);
    float*     denom     = (float*)alloc((size_t)N * 4);
    int*       counts    = (int*)alloc((size_t)R * 4);
    int*       offsets   = (int*)alloc((size_t)R * 4);
    int*       cursor    = (int*)alloc((size_t)R * 4);
    int*       sortedE   = (int*)alloc((size_t)E * 4);
    (void)ws_size; (void)in_sizes; (void)n_in;

    const int TB = 256;
    auto cdiv = [](int a, int b) { return (a + b - 1) / b; };

    // one-time: gather x, bucket edges by relation, zero att pad rows
    k_gather_x<<<cdiv(N * D, TB), TB, 0, stream>>>(entity, emb, xf, xh);
    k_zero_u32<<<cdiv(R, TB), TB, 0, stream>>>((unsigned*)counts, R);
    k_count<<<cdiv(E, TB), TB, 0, stream>>>(etype, counts);
    k_scan<<<1, 512, 0, stream>>>(counts, offsets, cursor);
    k_scatter<<<cdiv(E, TB), TB, 0, stream>>>(etype, cursor, sortedE);
    k_zero_u32<<<cdiv(NRELP * Bb / 2, TB), TB, 0, stream>>>((unsigned*)atth,
                                                            NRELP * Bb / 2);

    auto layer = [&](const float* xin, const _Float16* xinh, const float* basis,
                     const float* att, const float* wgt, const float* root,
                     const float* bias, float* out, int doRelu) {
        k_cast_f16<<<cdiv(R * Bb, TB), TB, 0, stream>>>(att, atth, R * Bb);
        k_tcast<<<cdiv(Bb * D * D, TB), TB, 0, stream>>>(basis, basisT, Bb, D * D);
        k_tcast<<<cdiv(D * D, TB), TB, 0, stream>>>(root, rootT, D, D);

        k_relW<<<dim3(NRELP / 16, (D * D) / 16), 32, 0, stream>>>(atth, basisT, Wf16);

        k_zero_u32<<<cdiv(N, TB), TB, 0, stream>>>(nodeKey, N);
        k_zero_u32<<<cdiv(N, TB), TB, 0, stream>>>((unsigned*)denom, N);
        k_alpha<<<cdiv(E * 32, 128), 128, 0, stream>>>(src, dst, etype, xin, wgt,
                                                       alphaRaw, nodeKey);
        k_expsum<<<cdiv(E, TB), TB, 0, stream>>>(dst, alphaRaw, nodeKey, aBuf, denom);
        k_scale<<<cdiv(E, TB), TB, 0, stream>>>(dst, aBuf, denom, scaleBuf);

        k_zero_u32<<<cdiv(N * D, TB), TB, 0, stream>>>((unsigned*)out, N * D);
        k_msg<<<R, 256, MSG_LDS_B, stream>>>(sortedE, offsets, counts, src, dst,
                                             xinh, Wf16, scaleBuf, out);
        k_root<<<128, 256, A_LDS_B, stream>>>(xinh, rootT, bias, out, doRelu);
    };

    layer(xf, xh, basis1, att1, wgt1, root1, bias1, hbuf, /*relu=*/1);
    k_cast_f16<<<cdiv(N * D, TB), TB, 0, stream>>>(hbuf, xh, N * D);
    layer(hbuf, xh, basis2, att2, wgt2, root2, bias2, (float*)d_out, /*relu=*/0);
    (void)out_size;
}
